// TNEPPerTypeANN_11338713661486
// MI455X (gfx1250) — matrix operand order repaired and checked by reading, not compiled, not run on hardware
//
#include <hip/hip_runtime.h>

// ---------------------------------------------------------------------------
// Problem constants (from reference): N=131072, DQ=256, T=4, H1=512
// ---------------------------------------------------------------------------
constexpr int DQ = 256;
constexpr int H1 = 512;
constexpr int TT = 4;
constexpr int TILE_M = 128;          // rows per block (8 waves x 16 rows)
constexpr int KT = DQ / 32;          // 8 k-tiles of 32
constexpr int NT = H1 / 16;          // 32 n-tiles of 16
constexpr int BTILE_DW = 8 * 32 * 8; // dwords per (t,nt) packed B tile = 2048 (8KB)

typedef __attribute__((ext_vector_type(16))) __bf16        v16bf;
typedef __attribute__((ext_vector_type(8)))  float         v8f;
typedef __attribute__((ext_vector_type(4)))  unsigned int  u32x4;
typedef __attribute__((ext_vector_type(8)))  int           i32x8;
typedef __attribute__((ext_vector_type(4)))  int           i32x4;

__device__ __forceinline__ unsigned int pk_bf16(float lo, float hi) {
  unsigned int ul = __float_as_uint(lo);
  unsigned int uh = __float_as_uint(hi);
  ul = (ul + 0x7FFFu + ((ul >> 16) & 1u)) >> 16;   // round-to-nearest-even
  uh = (uh + 0x7FFFu + ((uh >> 16) & 1u)) >> 16;
  return (ul & 0xFFFFu) | (uh << 16);
}

// Native gfx1250 v_tanh_f32 if the builtin exists on this toolchain;
// otherwise fall back to the ocml expansion.
__device__ __forceinline__ float fast_tanh(float x) {
#if __has_builtin(__builtin_amdgcn_tanhf)
  return __builtin_amdgcn_tanhf(x);
#else
  return tanhf(x);
#endif
}

// ---------------------------------------------------------------------------
// Pass 1: bin atoms by type
// ---------------------------------------------------------------------------
__global__ void zero_counts_kernel(int* counts) {
  if (threadIdx.x < TT) counts[threadIdx.x] = 0;
}

__global__ void bin_kernel(const int* __restrict__ Z, int* __restrict__ counts,
                           int* __restrict__ perm, int N) {
  int n = blockIdx.x * blockDim.x + threadIdx.x;
  if (n < N) {
    int t = Z[n];
    int p = atomicAdd(&counts[t], 1);
    perm[t * N + p] = n;
  }
}

// ---------------------------------------------------------------------------
// Pass 2: repack W0 (fp32 [T][DQ][H1]) -> bf16 in exact WMMA-B lane layout.
// Layout: dword index = ((t*NT + nt)*2048) + (kt*32 + lane)*8 + j
//   lane<16 : col = lane, Kbase = kt*32 + 0   ; lane>=16: Kbase = kt*32 + 8
//   dword j holds {K=Kbase+16*(j>>2)+2*(j&3) (lo16), +1 (hi16)} of column col.
// ---------------------------------------------------------------------------
__global__ void pack_w0_kernel(const float* __restrict__ W0,
                               unsigned int* __restrict__ out) {
  int tid  = blockIdx.x * blockDim.x + threadIdx.x;   // 262144 total
  int j    = tid & 7;
  int lane = (tid >> 3) & 31;
  int kt   = (tid >> 8) & 7;
  int nt   = (tid >> 11) & 31;
  int t    = (tid >> 16) & 3;
  int n    = nt * 16 + (lane & 15);
  int k0   = kt * 32 + ((lane >> 4) << 3) + ((j >> 2) << 4) + 2 * (j & 3);
  const float* w = W0 + ((size_t)t * DQ + k0) * H1 + n;
  out[tid] = pk_bf16(w[0], w[H1]);
}

// ---------------------------------------------------------------------------
// Pass 3: grouped GEMM + fused tanh epilogue + second-stage dot.
// One block = 128 gathered rows of one type; 8 waves x 16 rows each.
// B tiles (8KB) staged to LDS by the Tensor Data Mover, double buffered.
// ---------------------------------------------------------------------------
__device__ __forceinline__ void tdm_load_btile(const unsigned int* gsrc,
                                               unsigned int lds_byte_off) {
  unsigned long long ga = (unsigned long long)(__UINTPTR_TYPE__)gsrc;
  u32x4 g0;
  g0[0] = 1u;                                   // count=1, gather off
  g0[1] = lds_byte_off;                         // lds_addr (bytes)
  g0[2] = (unsigned int)ga;                     // global_addr[31:0]
  g0[3] = ((unsigned int)(ga >> 32) & 0x01FFFFFFu) | 0x80000000u; // addr[56:32] | type=2
  i32x8 g1;
  g1[0] = (int)(2u << 16);                      // data_size = 2 (4 bytes)
  g1[1] = (int)((unsigned)BTILE_DW << 16);      // tensor_dim0 lo = 2048 dwords
  g1[2] = (int)(1u << 16);                      // tensor_dim0 hi=0 | tensor_dim1=1
  g1[3] = (int)((unsigned)BTILE_DW << 16);      // tensor_dim1 hi=0 | tile_dim0=2048
  g1[4] = 1;                                    // tile_dim1=1 | tile_dim2=0
  g1[5] = BTILE_DW;                             // tensor_dim0_stride lo
  g1[6] = 0;
  g1[7] = 0;
  i32x4 gz4 = {0, 0, 0, 0};
  i32x8 gz8 = {0, 0, 0, 0, 0, 0, 0, 0};
  // amdgpu-toolchain (clang-23) 6-arg form: (g0, g1, g2, g3, extra_g, cpol)
  __builtin_amdgcn_tensor_load_to_lds(g0, g1, gz4, gz4, gz8, 0);
}

__global__ __launch_bounds__(256) void gemm_kernel(
    const float* __restrict__ q, const float* __restrict__ b0,
    const float* __restrict__ W1, const float* __restrict__ b1p,
    const int* __restrict__ counts, const int* __restrict__ perm,
    const unsigned int* __restrict__ packedB, float* __restrict__ F,
    int N, int numTiles) {
  __shared__ unsigned int ldsB[2 * BTILE_DW];   // 16 KB, double buffered, LDS off 0

  const int t    = blockIdx.x / numTiles;
  const int tile = blockIdx.x % numTiles;
  const int cnt  = counts[t];
  if (tile * TILE_M >= cnt) return;             // block-uniform early exit

  const int lane = threadIdx.x & 31;
  const int wave = threadIdx.x >> 5;
  const int lhalf = lane >> 4;                  // 0: lanes 0-15, 1: lanes 16-31
  const int lr    = lane & 15;

  const int rowBase = tile * TILE_M + wave * 16;
  int ridx = rowBase + lr;
  if (ridx > cnt - 1) ridx = cnt - 1;           // clamp (keeps EXEC full)
  const int row = perm[t * N + ridx];

  // ---- Build resident A tile (16 rows x 256 K) in bf16 WMMA layout --------
  union AV { v16bf v; unsigned int u[8]; };
  AV A[KT];
  const float* qr = q + (size_t)row * DQ;
#pragma unroll
  for (int kt = 0; kt < KT; ++kt) {
    int base = kt * 32 + lhalf * 8;
    float4 f0 = *(const float4*)(qr + base);
    float4 f1 = *(const float4*)(qr + base + 4);
    float4 f2 = *(const float4*)(qr + base + 16);
    float4 f3 = *(const float4*)(qr + base + 20);
    A[kt].u[0] = pk_bf16(f0.x, f0.y);  A[kt].u[1] = pk_bf16(f0.z, f0.w);
    A[kt].u[2] = pk_bf16(f1.x, f1.y);  A[kt].u[3] = pk_bf16(f1.z, f1.w);
    A[kt].u[4] = pk_bf16(f2.x, f2.y);  A[kt].u[5] = pk_bf16(f2.z, f2.w);
    A[kt].u[6] = pk_bf16(f3.x, f3.y);  A[kt].u[7] = pk_bf16(f3.z, f3.w);
  }

  float fpart[8] = {0.f, 0.f, 0.f, 0.f, 0.f, 0.f, 0.f, 0.f};

  // Prefetch first B tile via TDM (wave 0 issues; EXEC-independent DMA).
  if (wave == 0) {
    tdm_load_btile(packedB + (size_t)(t * NT + 0) * BTILE_DW, 0u);
  }

  for (int nt = 0; nt < NT; ++nt) {
    if (wave == 0) __builtin_amdgcn_s_wait_tensorcnt(0);
    __syncthreads();                            // buffer nt ready for all waves
    if (wave == 0 && nt + 1 < NT) {             // overlap DMA of nt+1 with compute
      tdm_load_btile(packedB + (size_t)(t * NT + nt + 1) * BTILE_DW,
                     (unsigned int)(((nt + 1) & 1) * BTILE_DW * 4));
    }

    const unsigned int* bb = &ldsB[(nt & 1) * BTILE_DW];
    v8f acc = {0.f, 0.f, 0.f, 0.f, 0.f, 0.f, 0.f, 0.f};
#pragma unroll
    for (int kt = 0; kt < KT; ++kt) {
      v16bf bv = *(const v16bf*)(bb + (kt * 32 + lane) * 8);
      acc = __builtin_amdgcn_wmma_f32_16x16x32_bf16(
          false, A[kt].v, false, bv, (short)0, acc, false, false);
    }

    // Fused epilogue: h = tanh(acc + b0[col]); fpart += h * W1[col]
    const int col = nt * 16 + lr;
    const float bcol = b0[t * H1 + col];
    const float wcol = W1[t * H1 + col];
#pragma unroll
    for (int g = 0; g < 8; ++g)
      fpart[g] += fast_tanh(acc[g] + bcol) * wcol;

    __syncthreads();                            // all reads done before nt+2 lands
  }

  // Reduce the 16 columns held across lanes of each half-wave.
#pragma unroll
  for (int g = 0; g < 8; ++g) {
    float v = fpart[g];
    v += __shfl_xor(v, 1, 32);
    v += __shfl_xor(v, 2, 32);
    v += __shfl_xor(v, 4, 32);
    v += __shfl_xor(v, 8, 32);
    fpart[g] = v;
  }

  if (lr == 0) {                                // lanes 0 and 16 hold row sums
    const float bias = b1p[0];
#pragma unroll
    for (int g = 0; g < 8; ++g) {
      int m = g + 8 * lhalf;                    // C layout: VGPR g -> M = g | 8+g
      int ri = rowBase + m;
      if (ri < cnt) {
        F[perm[t * N + ri]] = fpart[g] + bias;
      }
    }
  }
}

// ---------------------------------------------------------------------------
// Host launcher
// ---------------------------------------------------------------------------
extern "C" void kernel_launch(void* const* d_in, const int* in_sizes, int n_in,
                              void* d_out, int out_size, void* d_ws, size_t ws_size,
                              hipStream_t stream) {
  const float* q  = (const float*)d_in[0];
  const int*   Z  = (const int*)d_in[1];
  const float* W0 = (const float*)d_in[2];
  const float* b0 = (const float*)d_in[3];
  const float* W1 = (const float*)d_in[4];
  const float* b1 = (const float*)d_in[5];
  float* F = (float*)d_out;
  const int N = in_sizes[1];                    // length of Z

  // Workspace layout: counts @0, perm @256 (T*N ints), packed W0 @ 2MiB+4KiB
  int* counts = (int*)d_ws;
  int* perm   = (int*)((char*)d_ws + 256);
  const size_t packed_off = (size_t)256 + (size_t)TT * N * sizeof(int);
  const size_t packed_off_al = (packed_off + 4095) & ~(size_t)4095;
  unsigned int* packed = (unsigned int*)((char*)d_ws + packed_off_al);

  zero_counts_kernel<<<1, 64, 0, stream>>>(counts);
  bin_kernel<<<(N + 255) / 256, 256, 0, stream>>>(Z, counts, perm, N);

  const int packTotal = TT * NT * BTILE_DW;     // 262144 dwords
  pack_w0_kernel<<<packTotal / 256, 256, 0, stream>>>(W0, packed);

  const int numTiles = (N + TILE_M - 1) / TILE_M;
  gemm_kernel<<<TT * numTiles, 256, 0, stream>>>(q, b0, W1, b1, counts, perm,
                                                 packed, F, N, numTiles);
}
